// TransformerBlock_45260365365723
// MI455X (gfx1250) — matrix operand order repaired and checked by reading, not compiled, not run on hardware
//
#include <hip/hip_runtime.h>

// ---------------------------------------------------------------- types
typedef __bf16 bf16;
typedef __attribute__((ext_vector_type(16))) __bf16 v16bf;
typedef __attribute__((ext_vector_type(8)))  float  v8f;

static constexpr int Bn  = 2;
static constexpr int Tn  = 2048;
static constexpr int Dn  = 768;
static constexpr int Hn  = 12;
static constexpr int HDn = 64;
static constexpr int FFn = 3072;

__device__ __forceinline__ v16bf load_frag(const bf16* p0, const bf16* p1) {
  union { v16bf v; uint4 u[2]; } f;
  f.u[0] = *reinterpret_cast<const uint4*>(p0);
  f.u[1] = *reinterpret_cast<const uint4*>(p1);
  return f.v;
}

__device__ __forceinline__ v8f wmma_bf16(v16bf a, v16bf b, v8f c) {
  // D = A(16x32 bf16) * B(32x16 bf16) + C(16x16 f32)
  return __builtin_amdgcn_wmma_f32_16x16x32_bf16(
      /*neg_a=*/false, a, /*neg_b=*/false, b,
      /*c_mod=*/(short)0, c, /*reuse_a=*/false, /*reuse_b=*/false);
}

// CDNA5 async global->LDS copy, 16B per active lane, tracked by ASYNCcnt.
__device__ __forceinline__ void async_ld16(unsigned lds_byte_addr, const void* gaddr) {
  asm volatile("global_load_async_to_lds_b128 %0, %1, off"
               :: "v"(lds_byte_addr), "v"(gaddr) : "memory");
}
__device__ __forceinline__ void wait_async0() {
  asm volatile("s_wait_asynccnt 0" ::: "memory");
}
__device__ __forceinline__ unsigned lds_addr_of(const void* p) {
  return (unsigned)(uintptr_t)p;  // low 32 bits of flat shared addr = LDS offset
}

// ---------------------------------------------------------------- cast
__global__ __launch_bounds__(256) void cast_f32_bf16_kernel(
    const float* __restrict__ src, bf16* __restrict__ dst, int n) {
  int i = blockIdx.x * 256 + threadIdx.x;
  if (i < n) dst[i] = (bf16)src[i];
}

// ---------------------------------------------------------------- rmsnorm (fp32 in -> bf16 out)
__global__ __launch_bounds__(256) void rmsnorm_bf16_kernel(
    const float* __restrict__ x, const float* __restrict__ w,
    bf16* __restrict__ out) {
  const int row = blockIdx.x;
  const float* xr = x + (size_t)row * Dn;
  __shared__ float red[256];
  float s = 0.f;
  for (int c = threadIdx.x; c < Dn; c += 256) { float v = xr[c]; s += v * v; }
  red[threadIdx.x] = s;
  __syncthreads();
  for (int st = 128; st > 0; st >>= 1) {
    if (threadIdx.x < st) red[threadIdx.x] += red[threadIdx.x + st];
    __syncthreads();
  }
  const float norm  = sqrtf(red[0]);
  const float scale = 1.0f / (norm / sqrtf((float)Dn) + 1e-6f);
  bf16* orow = out + (size_t)row * Dn;
  for (int c = threadIdx.x; c < Dn; c += 256)
    orow[c] = (bf16)(w[c] * xr[c] * scale);
}

// ---------------------------------------------------------------- GEMM: C[M,N] = A[M,K] * W[N,K]^T
// Block tile 128x128, 8 waves (4 row x 2 col), wave tile 32x64 (8 WMMA / K-step).
// A/B K-panels (128x32 bf16) double-buffered in LDS, filled with
// global_load_async_to_lds_b128 (ASYNCcnt), overlapped with WMMA on other buffer.
static constexpr int LSTR = 40;  // LDS row stride in bf16 (80B): 20*i mod 64 -> no bank conflicts

__global__ __launch_bounds__(256) void gemm_bf16_kernel(
    const bf16* __restrict__ A, const bf16* __restrict__ W,
    int M, int N, int K,
    const float* __restrict__ res, float* __restrict__ outF,
    bf16* __restrict__ outB) {
  __shared__ __align__(16) bf16 As[2][128 * LSTR];
  __shared__ __align__(16) bf16 Bs[2][128 * LSTR];

  const int tid  = threadIdx.x;
  const int lane = tid & 31;
  const int wid  = tid >> 5;
  const int wrow = wid >> 1;          // 0..3  -> 32-row strip
  const int wcol = wid & 1;           // 0..1  -> 64-col strip
  const int row0 = blockIdx.y * 128;
  const int col0 = blockIdx.x * 128;
  if (row0 >= M || col0 >= N) return;

  // staging: tile = 128 rows x 32 cols = 512 chunks of 8 bf16 (16B); thread does 2 per matrix
  const int srow0 = tid >> 2;         // rows 0..63
  const int scc   = tid & 3;          // 16B chunk within row
  const int srow1 = srow0 + 64;       // rows 64..127

  const bf16* gA0 = A + (size_t)(row0 + srow0) * K + scc * 8;
  const bf16* gA1 = A + (size_t)(row0 + srow1) * K + scc * 8;
  const bf16* gB0 = W + (size_t)(col0 + srow0) * K + scc * 8;
  const bf16* gB1 = W + (size_t)(col0 + srow1) * K + scc * 8;
  const unsigned lA0 = lds_addr_of(&As[0][srow0 * LSTR + scc * 8]);
  const unsigned lA1 = lds_addr_of(&As[0][srow1 * LSTR + scc * 8]);
  const unsigned lB0 = lds_addr_of(&Bs[0][srow0 * LSTR + scc * 8]);
  const unsigned lB1 = lds_addr_of(&Bs[0][srow1 * LSTR + scc * 8]);
  const unsigned bufStrideA = (unsigned)(128 * LSTR * sizeof(bf16));

  auto stage = [&](int buf, int k0) {
    const unsigned bo = buf ? bufStrideA : 0u;
    async_ld16(lA0 + bo, gA0 + k0);
    async_ld16(lA1 + bo, gA1 + k0);
    async_ld16(lB0 + bo, gB0 + k0);
    async_ld16(lB1 + bo, gB1 + k0);
  };

  const int lm = lane & 15;
  const int lh = lane >> 4;

  v8f acc[2][4];
  const v8f vzero = {0.f, 0.f, 0.f, 0.f, 0.f, 0.f, 0.f, 0.f};
  #pragma unroll
  for (int mi = 0; mi < 2; ++mi)
    #pragma unroll
    for (int ni = 0; ni < 4; ++ni) acc[mi][ni] = vzero;

  const int nk = K / 32;
  stage(0, 0);
  wait_async0();
  __syncthreads();

  for (int ks = 0; ks < nk; ++ks) {
    const int buf = ks & 1;
    if (ks + 1 < nk) stage(buf ^ 1, (ks + 1) * 32);  // overlap copy with WMMA below

    v16bf af[2], bfm[4];
    #pragma unroll
    for (int mi = 0; mi < 2; ++mi) {
      const bf16* r = &As[buf][(wrow * 32 + 16 * mi + lm) * LSTR];
      af[mi] = load_frag(r + 8 * lh, r + 16 + 8 * lh);
    }
    #pragma unroll
    for (int ni = 0; ni < 4; ++ni) {
      const bf16* r = &Bs[buf][(wcol * 64 + 16 * ni + lm) * LSTR];
      bfm[ni] = load_frag(r + 8 * lh, r + 16 + 8 * lh);
    }
    #pragma unroll
    for (int mi = 0; mi < 2; ++mi)
      #pragma unroll
      for (int ni = 0; ni < 4; ++ni)
        acc[mi][ni] = wmma_bf16(af[mi], bfm[ni], acc[mi][ni]);

    wait_async0();     // next buffer landed (overlapped with the 8 WMMAs)
    __syncthreads();   // all waves done reading `buf` before it is restaged
  }

  // C layout: VGPR r, lane -> row = r + 8*(lane>>4), col = lane&15
  #pragma unroll
  for (int mi = 0; mi < 2; ++mi)
    #pragma unroll
    for (int ni = 0; ni < 4; ++ni)
      #pragma unroll
      for (int r = 0; r < 8; ++r) {
        const int rr = row0 + wrow * 32 + 16 * mi + r + 8 * lh;
        const int cc = col0 + wcol * 64 + 16 * ni + lm;
        const size_t idx = (size_t)rr * N + cc;
        float v = acc[mi][ni][r];
        if (outB) {
          outB[idx] = (bf16)v;
        } else {
          if (res) v += res[idx];
          outF[idx] = v;
        }
      }
}

// ---------------------------------------------------------------- RoPE + head permute (+ V transpose)
// q,k,v: bf16 [B,T,D];  qr,kr: bf16 [B,H,T,HD];  vt: bf16 [B,H,HD,T]
__global__ __launch_bounds__(256) void rope_permute_kernel(
    const bf16* __restrict__ q, const bf16* __restrict__ k,
    const bf16* __restrict__ v,
    const float* __restrict__ cosb, const float* __restrict__ sinb,
    bf16* __restrict__ qr, bf16* __restrict__ kr, bf16* __restrict__ vt) {
  const int idx = blockIdx.x * 256 + threadIdx.x;  // B*H*T*(HD/2)
  const int i    = idx & 31;
  const int t    = (idx >> 5) & (Tn - 1);
  const int rest = idx >> 16;            // 32*T == 65536
  const int hh   = rest % Hn;
  const int b    = rest / Hn;

  const size_t src = ((size_t)(b * Tn + t)) * Dn + hh * HDn;
  const float q0 = (float)q[src + 2 * i], q1 = (float)q[src + 2 * i + 1];
  const float k0 = (float)k[src + 2 * i], k1 = (float)k[src + 2 * i + 1];
  const float c = cosb[t * HDn + 2 * i];
  const float s = sinb[t * HDn + 2 * i];

  const size_t dst = (((size_t)(b * Hn + hh)) * Tn + t) * HDn + 2 * i;
  qr[dst]     = (bf16)(q0 * c - q1 * s);
  qr[dst + 1] = (bf16)(q1 * c + q0 * s);
  kr[dst]     = (bf16)(k0 * c - k1 * s);
  kr[dst + 1] = (bf16)(k1 * c + k0 * s);

  const size_t vb = ((size_t)(b * Hn + hh)) * HDn;
  vt[(vb + 2 * i) * Tn + t]     = v[src + 2 * i];
  vt[(vb + 2 * i + 1) * Tn + t] = v[src + 2 * i + 1];
}

// ---------------------------------------------------------------- flash attention (causal, wave-level)
// Q,K: [B,H,T,HD] bf16;  VT: [B,H,HD,T] bf16;  Y: merged [B,T,D] bf16
__global__ __launch_bounds__(256) void flash_attn_kernel(
    const bf16* __restrict__ Q, const bf16* __restrict__ Kb,
    const bf16* __restrict__ VT, bf16* __restrict__ Y) {
  __shared__ __align__(16) bf16 plds[8][16 * 32];  // per-wave P staging

  const int lane = threadIdx.x & 31;
  const int wid  = threadIdx.x >> 5;
  const int tile = blockIdx.x * 8 + wid;   // B*H*(T/16) tiles total
  const int qt   = tile & 127;             // T/16 = 128
  const int bh   = tile >> 7;
  const int hh   = bh % Hn;
  const int b    = bh / Hn;
  const int qbase = qt * 16;

  const bf16* qptr = Q  + (((size_t)(b * Hn + hh)) * Tn + qbase) * HDn;
  const bf16* kptr = Kb + (((size_t)(b * Hn + hh)) * Tn) * HDn;
  const bf16* vptr = VT + (((size_t)(b * Hn + hh)) * HDn) * Tn;

  const int lm = lane & 15;
  const int lh = lane >> 4;

  // Q A-frags: d-chunks {0..31}, {32..63}
  v16bf aq[2];
  #pragma unroll
  for (int dc = 0; dc < 2; ++dc)
    aq[dc] = load_frag(qptr + lm * HDn + 32 * dc + 8 * lh,
                       qptr + lm * HDn + 32 * dc + 16 + 8 * lh);

  const v8f vzero = {0.f, 0.f, 0.f, 0.f, 0.f, 0.f, 0.f, 0.f};
  v8f acc[4] = {vzero, vzero, vzero, vzero};  // y tile 16 x 64
  float mrow[8], lrow[8];
  #pragma unroll
  for (int r = 0; r < 8; ++r) { mrow[r] = -1e30f; lrow[r] = 0.f; }

  const int nkb = (qbase + 47) >> 5;  // 32-key blocks touching causal band
  for (int kb = 0; kb < nkb; ++kb) {
    const int kbase = kb * 32;
    // S = Q * K^T  (two 16x16 score frags, contraction over HD=64)
    v8f s0 = vzero, s1 = vzero;
    #pragma unroll
    for (int dc = 0; dc < 2; ++dc) {
      const bf16* k0p = kptr + (size_t)(kbase + lm) * HDn + 32 * dc + 8 * lh;
      const bf16* k1p = kptr + (size_t)(kbase + 16 + lm) * HDn + 32 * dc + 8 * lh;
      v16bf bk0 = load_frag(k0p, k0p + 16);
      v16bf bk1 = load_frag(k1p, k1p + 16);
      s0 = wmma_bf16(aq[dc], bk0, s0);
      s1 = wmma_bf16(aq[dc], bk1, s1);
    }

    const bool needmask = (kbase + 31) > qbase;
    #pragma unroll
    for (int r = 0; r < 8; ++r) {
      float v0 = s0[r] * 0.125f;   // 1/sqrt(HD)
      float v1 = s1[r] * 0.125f;
      if (needmask) {
        const int row = qbase + r + 8 * lh;
        if (kbase + lm > row)      v0 = -1e30f;
        if (kbase + 16 + lm > row) v1 = -1e30f;
      }
      // row max over 16 lanes of this half-wave
      float mx = fmaxf(v0, v1);
      #pragma unroll
      for (int off = 1; off < 16; off <<= 1)
        mx = fmaxf(mx, __shfl_xor(mx, off, 16));
      const float mnew = fmaxf(mrow[r], mx);
      const float p0 = __expf(v0 - mnew);
      const float p1 = __expf(v1 - mnew);
      float rs = p0 + p1;
      #pragma unroll
      for (int off = 1; off < 16; off <<= 1)
        rs += __shfl_xor(rs, off, 16);
      const float sf = __expf(mrow[r] - mnew);
      lrow[r] = lrow[r] * sf + rs;
      mrow[r] = mnew;
      // rescale running accumulators
      acc[0][r] *= sf; acc[1][r] *= sf; acc[2][r] *= sf; acc[3][r] *= sf;
      // stage P (C-layout) into wave-private LDS
      plds[wid][(r + 8 * lh) * 32 + lm]      = (bf16)p0;
      plds[wid][(r + 8 * lh) * 32 + 16 + lm] = (bf16)p1;
    }

    // reload P as a 16x32 A-fragment
    v16bf ap = load_frag(&plds[wid][lm * 32 + 8 * lh],
                         &plds[wid][lm * 32 + 16 + 8 * lh]);

    // PV: four 16x16 output frags over d
    #pragma unroll
    for (int fn = 0; fn < 4; ++fn) {
      const bf16* vp = vptr + (size_t)(16 * fn + lm) * Tn + kbase + 8 * lh;
      v16bf bv = load_frag(vp, vp + 16);
      acc[fn] = wmma_bf16(ap, bv, acc[fn]);
    }
  }

  // epilogue: y = acc / l, write merged [B,T,D] bf16
  #pragma unroll
  for (int fn = 0; fn < 4; ++fn)
    #pragma unroll
    for (int r = 0; r < 8; ++r) {
      const int trow = qbase + r + 8 * lh;
      const int d    = 16 * fn + lm;
      const float yv = acc[fn][r] / lrow[r];
      Y[((size_t)(b * Tn + trow)) * Dn + hh * HDn + d] = (bf16)yv;
    }
}

// ---------------------------------------------------------------- SiLU(a) * b  (bf16, in-place into a)
__global__ __launch_bounds__(256) void silu_mul_kernel(
    bf16* __restrict__ a, const bf16* __restrict__ b2, int n) {
  const int i = blockIdx.x * 256 + threadIdx.x;
  if (i >= n) return;
  const float av = (float)a[i];
  const float s  = av / (1.0f + __expf(-av));
  a[i] = (bf16)(s * (float)b2[i]);
}

// ---------------------------------------------------------------- host
extern "C" void kernel_launch(void* const* d_in, const int* in_sizes, int n_in,
                              void* d_out, int out_size, void* d_ws, size_t ws_size,
                              hipStream_t stream) {
  (void)in_sizes; (void)n_in; (void)out_size; (void)ws_size;
  const float* x    = (const float*)d_in[0];
  const float* wq   = (const float*)d_in[1];
  const float* wk   = (const float*)d_in[2];
  const float* wv   = (const float*)d_in[3];
  const float* wo   = (const float*)d_in[4];
  const float* w1   = (const float*)d_in[5];
  const float* w2   = (const float*)d_in[6];
  const float* w3   = (const float*)d_in[7];
  const float* anw  = (const float*)d_in[8];
  const float* fnw  = (const float*)d_in[9];
  const float* cosb = (const float*)d_in[10];
  const float* sinb = (const float*)d_in[11];
  float* out = (float*)d_out;

  // ---- workspace carve (aligned 256B)
  char* p = (char*)d_ws;
  auto alloc = [&](size_t elems, size_t esz) {
    void* r = (void*)p;
    p += (elems * esz + 255) & ~(size_t)255;
    return r;
  };
  const size_t nDD  = (size_t)Dn * Dn;
  const size_t nFD  = (size_t)FFn * Dn;
  const size_t nBTD = (size_t)Bn * Tn * Dn;
  const size_t nBTF = (size_t)Bn * Tn * FFn;

  bf16* wqb = (bf16*)alloc(nDD, 2);
  bf16* wkb = (bf16*)alloc(nDD, 2);
  bf16* wvb = (bf16*)alloc(nDD, 2);
  bf16* wob = (bf16*)alloc(nDD, 2);
  bf16* w1b = (bf16*)alloc(nFD, 2);
  bf16* w2b = (bf16*)alloc(nFD, 2);
  bf16* w3b = (bf16*)alloc(nFD, 2);
  bf16* xn   = (bf16*)alloc(nBTD, 2);
  bf16* qtmp = (bf16*)alloc(nBTD, 2);
  bf16* ktmp = (bf16*)alloc(nBTD, 2);
  bf16* vtmp = (bf16*)alloc(nBTD, 2);
  bf16* qr   = (bf16*)alloc(nBTD, 2);
  bf16* kr   = (bf16*)alloc(nBTD, 2);
  bf16* vt   = (bf16*)alloc(nBTD, 2);
  bf16* ybt  = (bf16*)alloc(nBTD, 2);
  float* x1  = (float*)alloc(nBTD, 4);
  bf16* hb   = (bf16*)alloc(nBTD, 2);
  bf16* ff1  = (bf16*)alloc(nBTF, 2);
  bf16* ff2  = (bf16*)alloc(nBTF, 2);

  auto cast = [&](const float* s, bf16* d, size_t n) {
    cast_f32_bf16_kernel<<<(int)((n + 255) / 256), 256, 0, stream>>>(s, d, (int)n);
  };
  auto gemm = [&](const bf16* A, const bf16* W, int M, int N, int K,
                  const float* res, float* oF, bf16* oB) {
    dim3 grid(N / 128, M / 128);
    gemm_bf16_kernel<<<grid, 256, 0, stream>>>(A, W, M, N, K, res, oF, oB);
  };

  const int Mrows = Bn * Tn;  // 4096

  // weights -> bf16
  cast(wq, wqb, nDD); cast(wk, wkb, nDD); cast(wv, wvb, nDD); cast(wo, wob, nDD);
  cast(w1, w1b, nFD); cast(w2, w2b, nFD); cast(w3, w3b, nFD);

  // attn pre-norm
  rmsnorm_bf16_kernel<<<Mrows, 256, 0, stream>>>(x, anw, xn);

  // QKV projections
  gemm(xn, wqb, Mrows, Dn, Dn, nullptr, nullptr, qtmp);
  gemm(xn, wkb, Mrows, Dn, Dn, nullptr, nullptr, ktmp);
  gemm(xn, wvb, Mrows, Dn, Dn, nullptr, nullptr, vtmp);

  // RoPE + head permute + V transpose
  {
    const int n = Bn * Hn * Tn * (HDn / 2);  // 1,572,864
    rope_permute_kernel<<<n / 256, 256, 0, stream>>>(qtmp, ktmp, vtmp, cosb, sinb,
                                                     qr, kr, vt);
  }

  // causal flash attention -> merged [B,T,D] bf16
  {
    const int tiles = Bn * Hn * (Tn / 16);   // 3072 wave tiles
    flash_attn_kernel<<<tiles / 8, 256, 0, stream>>>(qr, kr, vt, ybt);
  }

  // output projection + residual -> x1 (fp32)
  gemm(ybt, wob, Mrows, Dn, Dn, x, x1, nullptr);

  // ffn pre-norm
  rmsnorm_bf16_kernel<<<Mrows, 256, 0, stream>>>(x1, fnw, hb);

  // FFN
  gemm(hb, w1b, Mrows, FFn, Dn, nullptr, nullptr, ff1);
  gemm(hb, w2b, Mrows, FFn, Dn, nullptr, nullptr, ff2);
  silu_mul_kernel<<<(int)(nBTF / 256), 256, 0, stream>>>(ff1, ff2, (int)nBTF);
  gemm(ff1, w3b, Mrows, Dn, FFn, x1, out, nullptr);
}
